// token_input_embedder_module_13039520711505
// MI455X (gfx1250) — matrix operand order repaired and checked by reading, not compiled, not run on hardware
//
#include <hip/hip_runtime.h>
#include <hip/hip_bf16.h>
#include <math.h>

// ---------------- CDNA5 WMMA types ----------------
typedef __attribute__((ext_vector_type(16))) _Float16 v16h;
typedef __attribute__((ext_vector_type(8)))  float    v8f;

#define N_ATOMS 11776
#define NBLK    368
#define N_TOK   512

// fast reciprocal / sigmoid: v_rcp_f32 instead of IEEE div chain
__device__ __forceinline__ float frcp(float x) { return __builtin_amdgcn_rcpf(x); }
__device__ __forceinline__ float sigmf(float x) { return frcp(1.f + __expf(-x)); }

template<bool ARELU>
__device__ __forceinline__ void pack4(v16h& f, int base, float4 v)
{
  if (ARELU) {
    v.x = fmaxf(v.x, 0.f); v.y = fmaxf(v.y, 0.f);
    v.z = fmaxf(v.z, 0.f); v.w = fmaxf(v.w, 0.f);
  }
  f[base + 0] = (_Float16)v.x; f[base + 1] = (_Float16)v.y;
  f[base + 2] = (_Float16)v.z; f[base + 3] = (_Float16)v.w;
}

// ============================================================================
// Generic WMMA GEMM:  C[M,N] = epilogue( act( A' @ W^T + bias ) )
//   K, ARELU compile-time: straight-line v_wmma_f32_16x16x32_f16 chain,
//   b128 fragment loads, no in-loop branches.
//   amode 1 = analytic k_idx gather (row r -> clamp(r/128*32 - 48 + r%128)).
//   act: 0 none, 1 relu, 2 sigmoid, 3 silu.  mulsrc: r*=..., addsrc: r+=...
//   pairu: fused outer-sum epilogue for the 512x512x128 pair output,
//          r += u[i,c] + u[j,128+c], stored non-temporally (write-once 134MB).
// ============================================================================
template<int K, bool ARELU>
__global__ void wmma_gemm_kernel(const float* __restrict__ A, int lda,
                                 const float* __restrict__ W,
                                 float* __restrict__ C,
                                 const float* __restrict__ bias,
                                 const float* __restrict__ mulsrc,
                                 const float* __restrict__ addsrc,
                                 const float* __restrict__ pairu,
                                 int M, int N, int amode, int act)
{
  int lane = threadIdx.x & 31;
  int wave = threadIdx.x >> 5;
  int Nt = N >> 4;
  long tiles = (long)(M >> 4) * Nt;
  long tile  = (long)blockIdx.x * (blockDim.x >> 5) + wave;
  if (tile >= tiles) return;
  int tm = (int)(tile / Nt), tn = (int)(tile % Nt);
  int row0 = tm << 4, col0 = tn << 4;

  int mL = lane & 15;     // M row (A/D) or N col (B/D) within tile
  int hi = lane >> 4;     // lane half selects K sub-range

  int aRow = row0 + mL;
  int ar = aRow;
  if (amode == 1) {
    int nb = aRow >> 7, j = aRow & 127;
    int w = nb * 32 - 48 + j;
    ar = w < 0 ? 0 : (w > (N_ATOMS - 1) ? (N_ATOMS - 1) : w);
  }
  const float* Arow  = A + (long)ar * lda;
  const float* Wrow0 = W + (long)(col0 + mL) * K + hi * 16;

  v8f acc = {};
#pragma unroll 2
  for (int k0 = 0; k0 < K; k0 += 32) {
    v16h af, bf;
    // A fragment (16x32 f16): lanes<16 take K {0..15}, lanes>=16 K {16..31}.
    const float* ap = Arow + k0 + hi * 16;
    const float4* ap4  = reinterpret_cast<const float4*>(ap);
    const float4* ap4b = reinterpret_cast<const float4*>(ap + 8);
    pack4<ARELU>(af, 0,  ap4[0]);
    pack4<ARELU>(af, 4,  ap4[1]);
    pack4<ARELU>(af, 8,  ap4b[0]);
    pack4<ARELU>(af, 12, ap4b[1]);
    // B fragment (32x16 f16): B[k][n] = W[n][k]; lane = n, K half = hi*16.
    const float4* wp4 = reinterpret_cast<const float4*>(Wrow0 + k0);
    pack4<false>(bf, 0,  wp4[0]);
    pack4<false>(bf, 4,  wp4[1]);
    pack4<false>(bf, 8,  wp4[2]);
    pack4<false>(bf, 12, wp4[3]);
    acc = __builtin_amdgcn_wmma_f32_16x16x32_f16(false, af, false, bf,
                                                 (short)0, acc, false, false);
  }

  // D: lanes 0-15 -> M=v, lanes 16-31 -> M=v+8; N = mL
  int n = col0 + mL;
  float bv = bias ? bias[n] : 0.f;
#pragma unroll
  for (int v = 0; v < 8; ++v) {
    long m   = row0 + hi * 8 + v;
    long idx = m * (long)N + n;
    float r = acc[v] + bv;
    if      (act == 1) r = fmaxf(r, 0.f);
    else if (act == 2) r = sigmf(r);
    else if (act == 3) r = r * sigmf(r);
    if (mulsrc) r *= mulsrc[idx];
    if (addsrc) r += addsrc[idx];
    if (pairu) {
      int ti = (int)(m >> 9);          // token i (row of 512x512 pair grid)
      int tj = (int)(m & (N_TOK - 1)); // token j
      r += pairu[(long)ti * 256 + n] + pairu[(long)tj * 256 + 128 + n];
      __builtin_nontemporal_store(r, &C[idx]);   // write-once 134MB stream
    } else {
      C[idx] = r;
    }
  }
}

// ---------------- LayerNorm over 128 (no affine) ----------------
__global__ void ln128_kernel(const float* __restrict__ in, float* __restrict__ out)
{
  __shared__ float red[128];
  int row = blockIdx.x, c = threadIdx.x;
  float x = in[(long)row * 128 + c];
  red[c] = x; __syncthreads();
  for (int s = 64; s > 0; s >>= 1) { if (c < s) red[c] += red[c + s]; __syncthreads(); }
  float mu = red[0] * (1.f / 128.f); __syncthreads();
  float d = x - mu;
  red[c] = d * d; __syncthreads();
  for (int s = 64; s > 0; s >>= 1) { if (c < s) red[c] += red[c + s]; __syncthreads(); }
  float rs = rsqrtf(red[0] * (1.f / 128.f) + 1e-5f);
  out[(long)row * 128 + c] = d * rs;
}

// x = sigmoid(gb[:,:128]) * ln(a) + gb[:,128:]
__global__ void modulate_kernel(const float* __restrict__ gb,
                                const float* __restrict__ a,
                                float* __restrict__ x)
{
  __shared__ float red[128];
  int row = blockIdx.x, c = threadIdx.x;
  float av = a[(long)row * 128 + c];
  red[c] = av; __syncthreads();
  for (int s = 64; s > 0; s >>= 1) { if (c < s) red[c] += red[c + s]; __syncthreads(); }
  float mu = red[0] * (1.f / 128.f); __syncthreads();
  float d = av - mu;
  red[c] = d * d; __syncthreads();
  for (int s = 64; s > 0; s >>= 1) { if (c < s) red[c] += red[c + s]; __syncthreads(); }
  float lnv = d * rsqrtf(red[0] * (1.f / 128.f) + 1e-5f);
  float g0 = gb[(long)row * 256 + c];
  float g1 = gb[(long)row * 256 + 128 + c];
  x[(long)row * 128 + c] = sigmf(g0) * lnv + g1;
}

// ---------------- fused pair-bias: p=ph+pw; p+=MLP(p); LN; 16->4 ----------------
// grid = N_ATOMS (= nb*32+q), block = 128 (one thread per k)
__global__ void pairbias_kernel(const float* __restrict__ ph, const float* __restrict__ pw,
                                const float* __restrict__ W1, const float* __restrict__ W2,
                                const float* __restrict__ lnw, const float* __restrict__ lnb,
                                const float* __restrict__ pbw, float* __restrict__ pb)
{
  __shared__ float sW1[256], sW2[256], sph[16], slnw[16], slnb[16], spbw[64];
  int tid = threadIdx.x;
  int nb = blockIdx.x >> 5, q = blockIdx.x & 31;
  for (int i = tid; i < 256; i += 128) { sW1[i] = W1[i]; sW2[i] = W2[i]; }
  if (tid < 16) { sph[tid] = ph[(long)blockIdx.x * 16 + tid];
                  slnw[tid] = lnw[tid]; slnb[tid] = lnb[tid]; }
  if (tid < 64) spbw[tid] = pbw[tid];
  __syncthreads();

  int k = tid;
  const float* pwr = pw + ((long)nb * 128 + k) * 16;
  float p[16], t[16], m1[16];
#pragma unroll
  for (int i = 0; i < 16; ++i) { p[i] = sph[i] + pwr[i]; t[i] = fmaxf(p[i], 0.f); }
#pragma unroll
  for (int i = 0; i < 16; ++i) {
    float s = 0.f;
#pragma unroll
    for (int j = 0; j < 16; ++j) s += sW1[i * 16 + j] * t[j];
    m1[i] = fmaxf(s, 0.f);
  }
#pragma unroll
  for (int i = 0; i < 16; ++i) {
    float s = 0.f;
#pragma unroll
    for (int j = 0; j < 16; ++j) s += sW2[i * 16 + j] * m1[j];
    p[i] += s;
  }
  float mu = 0.f;
#pragma unroll
  for (int i = 0; i < 16; ++i) mu += p[i];
  mu *= (1.f / 16.f);
  float var = 0.f;
#pragma unroll
  for (int i = 0; i < 16; ++i) { float d = p[i] - mu; var += d * d; }
  float rs = rsqrtf(var * (1.f / 16.f) + 1e-5f);
  float y[16];
#pragma unroll
  for (int i = 0; i < 16; ++i) y[i] = (p[i] - mu) * rs * slnw[i] + slnb[i];
#pragma unroll
  for (int h = 0; h < 4; ++h) {
    float s = 0.f;
#pragma unroll
    for (int i = 0; i < 16; ++i) s += spbw[h * 16 + i] * y[i];
    pb[(((long)(nb * 4 + h) * 32 + q) * 128) + k] = s;  // [nb][h][q][k]
  }
}

// ---------------- local attention: one wave32 per (nb, head) ----------------
__global__ void __launch_bounds__(32)
attn_kernel(const float* __restrict__ qkv, const float* __restrict__ pb,
            const float* __restrict__ qbias, float* __restrict__ o)
{
  __shared__ float kS[128 * 32], vS[128 * 32], sS[32 * 128];
  int nb = blockIdx.x >> 2, h = blockIdx.x & 3;
  int lane = threadIdx.x;
  for (int r = lane; r < 128; r += 32) {
    int w = nb * 32 - 48 + r;
    int kr = w < 0 ? 0 : (w > (N_ATOMS - 1) ? (N_ATOMS - 1) : w);
    const float* src = qkv + (long)kr * 384;
#pragma unroll
    for (int d = 0; d < 32; ++d) {
      kS[r * 32 + d] = src[128 + h * 32 + d];
      vS[r * 32 + d] = src[256 + h * 32 + d];
    }
  }
  __syncthreads();
  int q = lane;
  int arow = nb * 32 + q;
  float qreg[32];
  const float* qsrc = qkv + (long)arow * 384 + h * 32;
#pragma unroll
  for (int d = 0; d < 32; ++d) qreg[d] = qsrc[d] + qbias[h * 32 + d];
  const float scale = 0.1767766952966369f; // 1/sqrt(32)
  const float* pbrow = pb + (((long)(nb * 4 + h) * 32 + q) * 128);
  float mx = -3.0e38f;
  for (int k = 0; k < 128; ++k) {
    float s = 0.f;
#pragma unroll
    for (int d = 0; d < 32; ++d) s += qreg[d] * kS[k * 32 + d];
    s = s * scale + pbrow[k];
    int w = nb * 32 - 48 + k;
    if (w < 0 || w >= N_ATOMS) s += -1.0e9f;
    sS[q * 128 + k] = s;
    mx = fmaxf(mx, s);
  }
  float den = 0.f;
  for (int k = 0; k < 128; ++k) { float e = __expf(sS[q * 128 + k] - mx); sS[q * 128 + k] = e; den += e; }
  float inv = frcp(den);
  float oreg[32];
#pragma unroll
  for (int d = 0; d < 32; ++d) oreg[d] = 0.f;
  for (int k = 0; k < 128; ++k) {
    float wgt = sS[q * 128 + k];
#pragma unroll
    for (int d = 0; d < 32; ++d) oreg[d] += wgt * vS[k * 32 + d];
  }
  float* od = o + (long)arow * 128 + h * 32;
#pragma unroll
  for (int d = 0; d < 32; ++d) od[d] = oreg[d] * inv;
}

// ---------------- small elementwise / pooling kernels ----------------
__global__ void zero_kernel(float* p, long n)
{ long i = (long)blockIdx.x * blockDim.x + threadIdx.x; if (i < n) p[i] = 0.f; }

__global__ void gateupd_kernel(float* a, const float* gate, const float* h, long n)
{ long i = (long)blockIdx.x * blockDim.x + threadIdx.x; if (i < n) a[i] += gate[i] * h[i]; }

__global__ void scatter_kernel(const float* __restrict__ atom_out, const int* __restrict__ a2t,
                               float* __restrict__ sums, float* __restrict__ cnt)
{
  int i = blockIdx.x, c = threadIdx.x;
  int t = a2t[i];
  atomicAdd(&sums[(long)t * 384 + c], atom_out[(long)i * 384 + c]);
  if (c == 0) atomicAdd(&cnt[t], 1.f);
}

__global__ void tokmean_kernel(const float* sums, const float* cnt, float* tok)
{
  int t = blockIdx.x, c = threadIdx.x;
  tok[(long)t * 384 + c] = sums[(long)t * 384 + c] * frcp(fmaxf(cnt[t], 1.f));
}

// ---------------- host-side launch helpers ----------------
static void gemm(hipStream_t s, const float* A, int lda, const float* W, float* C,
                 const float* bias, const float* mulsrc, const float* addsrc,
                 const float* pairu,
                 int M, int N, int K, int amode, int arelu, int act)
{
  long tiles = (long)(M / 16) * (N / 16);
  dim3 g((unsigned)((tiles + 7) / 8)), b(256);
#define LAUNCH(KK, RR) wmma_gemm_kernel<KK, RR><<<g, b, 0, s>>>( \
      A, lda, W, C, bias, mulsrc, addsrc, pairu, M, N, amode, act)
  if (arelu) {
    LAUNCH(128, true);
  } else if (K == 32) {
    LAUNCH(32, false);
  } else if (K == 256) {
    LAUNCH(256, false);
  } else if (K == 384) {
    LAUNCH(384, false);
  } else {
    LAUNCH(128, false);
  }
#undef LAUNCH
}

extern "C" void kernel_launch(void* const* d_in, const int* in_sizes, int n_in,
                              void* d_out, int out_size, void* d_ws, size_t ws_size,
                              hipStream_t stream)
{
  (void)in_sizes; (void)n_in; (void)out_size; (void)ws_size;
  const float* atom_feats = (const float*)d_in[0];
  const float* tpf        = (const float*)d_in[1];
  const float* W_cond     = (const float*)d_in[2];
  const float* W_ph       = (const float*)d_in[3];
  const float* W_pw       = (const float*)d_in[4];
  const float* W_mlp1     = (const float*)d_in[5];
  const float* W_mlp2     = (const float*)d_in[6];
  const float* pb_ln_w    = (const float*)d_in[7];
  const float* pb_ln_b    = (const float*)d_in[8];
  const float* pb_w       = (const float*)d_in[9];
  const float* sln_w      = (const float*)d_in[10];
  const float* qkv_w      = (const float*)d_in[11];
  const float* q_bias     = (const float*)d_in[12];
  const float* out_w      = (const float*)d_in[13];
  const float* out_b      = (const float*)d_in[14];
  const float* ada_w      = (const float*)d_in[15];
  const float* ta_w       = (const float*)d_in[16];
  const float* tb_w       = (const float*)d_in[17];
  const float* gate_w     = (const float*)d_in[18];
  const float* gate_b     = (const float*)d_in[19];
  const float* W_tok      = (const float*)d_in[20];
  const float* W_trunk    = (const float*)d_in[21];
  const float* W_struct   = (const float*)d_in[22];
  const float* W_pairin   = (const float*)d_in[23];
  const float* W_outer    = (const float*)d_in[24];
  // d_in[25..27] (q_idx / k_idx / k_mask) are analytic; recomputed on device.
  const int* a2t          = (const int*)d_in[28];

  float* ws = (float*)d_ws;
  size_t off = 0;
  float* A    = ws + off; off += (size_t)N_ATOMS * 128;       // current residual 'a'
  float* SLN  = ws + off; off += (size_t)N_ATOMS * 128;       // ln(c), fixed
  float* GB   = ws + off; off += (size_t)N_ATOMS * 256;       // gb / g (SwiGLU)
  float* X    = ws + off; off += (size_t)N_ATOMS * 128;       // x / xt / gate
  float* QKV  = ws + off; off += (size_t)N_ATOMS * 384;       // qkv / atom_out
  float* O    = ws + off; off += (size_t)N_ATOMS * 128;       // attn out / h
  float* PB   = ws + off; off += (size_t)NBLK * 4 * 32 * 128; // pair bias
  float* PH   = ws + off; off += (size_t)N_ATOMS * 16;
  float* PW   = ws + off; off += (size_t)NBLK * 128 * 16;
  float* SUMS = ws + off; off += (size_t)N_TOK * 384;
  float* CNT  = ws + off; off += (size_t)N_TOK;               // contiguous after SUMS
  float* TOK  = ws + off; off += (size_t)N_TOK * 384;
  float* U    = ws + off; off += (size_t)N_TOK * 256;

  float* out_trunk  = (float*)d_out;
  float* out_struct = out_trunk + (size_t)N_TOK * 384;
  float* out_pair   = out_struct + (size_t)N_TOK * 384;

  // c = atom_feats @ W_cond.T ; s_ln = ln(c)
  gemm(stream, atom_feats, 128, W_cond, A, nullptr, nullptr, nullptr, nullptr,
       N_ATOMS, 128, 128, 0, 0, 0);
  ln128_kernel<<<N_ATOMS, 128, 0, stream>>>(A, SLN);

  // ph = relu(s_ln) @ W_ph.T ; pw = relu(s_ln[k_idx]) @ W_pw.T (gather fused)
  gemm(stream, SLN, 128, W_ph, PH, nullptr, nullptr, nullptr, nullptr,
       N_ATOMS, 16, 128, 0, 1, 0);
  gemm(stream, SLN, 128, W_pw, PW, nullptr, nullptr, nullptr, nullptr,
       NBLK * 128, 16, 128, 1, 1, 0);
  pairbias_kernel<<<N_ATOMS, 128, 0, stream>>>(PH, PW, W_mlp1, W_mlp2,
                                               pb_ln_w, pb_ln_b, pb_w, PB);

  for (int l = 0; l < 3; ++l) {
    // adaLN modulation + qkv
    gemm(stream, SLN, 128, sln_w + (size_t)l * 256 * 128, GB,
         nullptr, nullptr, nullptr, nullptr, N_ATOMS, 256, 128, 0, 0, 0);
    modulate_kernel<<<N_ATOMS, 128, 0, stream>>>(GB, A, X);
    gemm(stream, X, 128, qkv_w + (size_t)l * 384 * 128, QKV,
         nullptr, nullptr, nullptr, nullptr, N_ATOMS, 384, 128, 0, 0, 0);
    // local biased attention
    attn_kernel<<<NBLK * 4, 32, 0, stream>>>(QKV, PB, q_bias + (size_t)l * 128, O);
    // a = a + o @ out_w.T + out_b   (residual fused, in place)
    gemm(stream, O, 128, out_w + (size_t)l * 128 * 128, A,
         out_b + (size_t)l * 128, nullptr, A, nullptr, N_ATOMS, 128, 128, 0, 0, 0);
    // transition: xt, SwiGLU (silu half in epilogue, mul fused in 2nd GEMM)
    gemm(stream, SLN, 128, ada_w + (size_t)l * 256 * 128, GB,
         nullptr, nullptr, nullptr, nullptr, N_ATOMS, 256, 128, 0, 0, 0);
    modulate_kernel<<<N_ATOMS, 128, 0, stream>>>(GB, A, X);
    gemm(stream, X, 128, ta_w + (size_t)l * 512 * 128, GB,
         nullptr, nullptr, nullptr, nullptr, N_ATOMS, 256, 128, 0, 0, 3); // silu(ab1)
    gemm(stream, X, 128, ta_w + (size_t)l * 512 * 128 + (size_t)256 * 128, GB,
         nullptr, GB, nullptr, nullptr, N_ATOMS, 256, 128, 0, 0, 0);      // *ab2
    gemm(stream, GB, 256, tb_w + (size_t)l * 128 * 256, O,
         nullptr, nullptr, nullptr, nullptr, N_ATOMS, 128, 256, 0, 0, 0); // h
    gemm(stream, SLN, 128, gate_w + (size_t)l * 128 * 128, X,
         gate_b + (size_t)l * 128, nullptr, nullptr, nullptr,
         N_ATOMS, 128, 128, 0, 0, 2);                                     // gate
    gateupd_kernel<<<(N_ATOMS * 128 + 255) / 256, 256, 0, stream>>>(
        A, X, O, (long)N_ATOMS * 128);
  }

  // atom_out = relu(a @ W_tok.T); segment mean -> tok
  gemm(stream, A, 128, W_tok, QKV, nullptr, nullptr, nullptr, nullptr,
       N_ATOMS, 384, 128, 0, 0, 1);
  long zn = (long)N_TOK * 384 + N_TOK;
  zero_kernel<<<(int)((zn + 255) / 256), 256, 0, stream>>>(SUMS, zn);
  scatter_kernel<<<N_ATOMS, 384, 0, stream>>>(QKV, a2t, SUMS, CNT);
  tokmean_kernel<<<N_TOK, 384, 0, stream>>>(SUMS, CNT, TOK);

  // output projections
  gemm(stream, TOK, 384, W_trunk, out_trunk, nullptr, nullptr, nullptr, nullptr,
       N_TOK, 384, 384, 0, 0, 0);
  gemm(stream, TOK, 384, W_struct, out_struct, nullptr, nullptr, nullptr, nullptr,
       N_TOK, 384, 384, 0, 0, 0);
  gemm(stream, TOK, 384, W_outer, U, nullptr, nullptr, nullptr, nullptr,
       N_TOK, 256, 384, 0, 0, 0);
  // pair = tpf @ W_pairin.T + u[i,:128] + u[j,128:], fused, NT stores to d_out
  gemm(stream, tpf, 32, W_pairin, out_pair, nullptr, nullptr, nullptr, U,
       N_TOK * N_TOK, 128, 32, 0, 0, 0);
}
